// Video_Gandalf_Model_V1_57466662420611
// MI455X (gfx1250) — compile-verified, matrix operand order checked by reading, not run on hardware
//
#include <hip/hip_runtime.h>
#include <hip/hip_bf16.h>

typedef _Float16 half8 __attribute__((ext_vector_type(8)));
typedef _Float16 v16h  __attribute__((ext_vector_type(16)));
typedef float    v8f   __attribute__((ext_vector_type(8)));

#define B_ROWS 4096
#define F_FEAT 256
#define H_DIM  64
#define NBK    8
#define D_DIM  128
#define KP     (F_FEAT * NBK)   // 2048 : inner dim of GEMM1
#define NZ     256              // hidden width / inner dim of GEMM2
#define LLMW   4096             // output width

// ---------------------------------------------------------------------------
// Kernel 1: EW[f,n,j] = sum_d emb[f,n,d] * proj_w1[f*128+d, j]
// stored transposed (f16) as ewt[j][f*8+n]  -> contiguous K for WMMA B-frags.
// ---------------------------------------------------------------------------
__global__ __launch_bounds__(256) void k_ew_prep(const float* __restrict__ emb,
                                                 const float* __restrict__ pw1,
                                                 _Float16* __restrict__ ewt) {
    __shared__ float embs[NBK * D_DIM];   // 4 KB: emb[f] tile
    const int f   = blockIdx.x;
    const int tid = threadIdx.x;
    for (int i = tid; i < NBK * D_DIM; i += 256)
        embs[i] = emb[(size_t)f * NBK * D_DIM + i];
    __syncthreads();

    const int j = tid;                    // output column 0..255
    float acc[NBK];
#pragma unroll
    for (int n = 0; n < NBK; ++n) acc[n] = 0.f;
    const float* wp = pw1 + (size_t)f * D_DIM * 256 + j;
    for (int d = 0; d < D_DIM; ++d) {
        float w = wp[(size_t)d * 256];    // coalesced across j
#pragma unroll
        for (int n = 0; n < NBK; ++n) acc[n] += embs[n * D_DIM + d] * w;
    }
    half8 o;
#pragma unroll
    for (int n = 0; n < NBK; ++n) o[n] = (_Float16)acc[n];
    *reinterpret_cast<half8*>(ewt + (size_t)j * KP + f * NBK) = o;  // b128 store
}

// ---------------------------------------------------------------------------
// Kernel 2: transpose proj_w2 [256][4096] f32 -> w2t [l][k] f16 via LDS tiles.
// ---------------------------------------------------------------------------
__global__ __launch_bounds__(256) void k_w2t(const float* __restrict__ w2,
                                             _Float16* __restrict__ w2t) {
    __shared__ float tile[32][33];
    const int l0 = blockIdx.x * 32;
    const int k0 = blockIdx.y * 32;
    const int tx = threadIdx.x;  // 0..31
    const int ty = threadIdx.y;  // 0..7
#pragma unroll
    for (int i = 0; i < 4; ++i)
        tile[ty + i * 8][tx] = w2[(size_t)(k0 + ty + i * 8) * LLMW + l0 + tx];
    __syncthreads();
#pragma unroll
    for (int i = 0; i < 4; ++i)
        w2t[(size_t)(l0 + ty + i * 8) * NZ + k0 + tx] = (_Float16)tile[tx][ty + i * 8];
}

// ---------------------------------------------------------------------------
// Kernel 3: fused sanitize -> linear1(sum of 3 copies) -> leaky -> linear2
//           -> leaky*tau -> softmax ; writes p f16 [B][F*8].
// One block per feature f; per-f weights staged in LDS.
// ---------------------------------------------------------------------------
__global__ __launch_bounds__(256) void k_probs(const float* __restrict__ stat,
                                               const float* __restrict__ l1w,
                                               const float* __restrict__ l1b,
                                               const float* __restrict__ l2w,
                                               const float* __restrict__ l2b,
                                               const float* __restrict__ tau,
                                               _Float16* __restrict__ pbuf) {
    __shared__ float s_w1s[H_DIM];
    __shared__ float s_b1[H_DIM];
    __shared__ float s_w2[H_DIM * NBK];
    __shared__ float s_b2[NBK];
    __shared__ float s_tau[NBK];
    const int f   = blockIdx.x;
    const int tid = threadIdx.x;

    if (tid < H_DIM) {
        // linear1_w is [F,3,H]; x is replicated 3x -> sum the 3 slices
        s_w1s[tid] = l1w[(size_t)(f * 3 + 0) * H_DIM + tid]
                   + l1w[(size_t)(f * 3 + 1) * H_DIM + tid]
                   + l1w[(size_t)(f * 3 + 2) * H_DIM + tid];
        s_b1[tid]  = l1b[(size_t)f * H_DIM + tid];
    }
    for (int i = tid; i < H_DIM * NBK; i += 256)
        s_w2[i] = l2w[(size_t)f * H_DIM * NBK + i];
    if (tid < NBK) { s_b2[tid] = l2b[f * NBK + tid]; s_tau[tid] = tau[f * NBK + tid]; }
    __syncthreads();

    for (int b = tid; b < B_ROWS; b += 256) {
        float x = stat[(size_t)b * F_FEAT + f];
        x = (x != x) ? 0.f : x;                 // nan -> 0
        x = fminf(fmaxf(x, 0.f), 65536.f);      // clamp

        float g[NBK];
#pragma unroll
        for (int n = 0; n < NBK; ++n) g[n] = s_b2[n];
        for (int hh = 0; hh < H_DIM; ++hh) {
            float a = x * s_w1s[hh] + s_b1[hh];
            float h = (a > 0.f) ? a : 0.01f * a;   // leaky relu
#pragma unroll
            for (int n = 0; n < NBK; ++n) g[n] += h * s_w2[hh * NBK + n];
        }
        float m = -3.4e38f;
#pragma unroll
        for (int n = 0; n < NBK; ++n) {
            float v = g[n];
            v = (v > 0.f) ? v : 0.01f * v;       // leaky relu
            v *= s_tau[n];                       // tau scale
            g[n] = v;
            m = fmaxf(m, v);
        }
        float sum = 0.f;
#pragma unroll
        for (int n = 0; n < NBK; ++n) { float e = __expf(g[n] - m); g[n] = e; sum += e; }
        float inv = 1.f / sum;
        half8 o;
#pragma unroll
        for (int n = 0; n < NBK; ++n) o[n] = (_Float16)(g[n] * inv);
        *reinterpret_cast<half8*>(pbuf + (size_t)b * KP + f * NBK) = o;  // b128 store
    }
}

// ---------------------------------------------------------------------------
// WMMA helpers (CDNA5 wave32 16x16x32 f16 fragment layout):
// lane = 16*kh + idx ; A: row=idx, B: col=idx ; K chunk = kh*8 + {0..7,16..23}
// ---------------------------------------------------------------------------
__device__ __forceinline__ v16h ld_frag(const _Float16* row, int k0, int kh) {
    half8 lo = *reinterpret_cast<const half8*>(row + k0 + kh * 8);
    half8 hi = *reinterpret_cast<const half8*>(row + k0 + 16 + kh * 8);
    v16h r;
#pragma unroll
    for (int i = 0; i < 8; ++i) { r[i] = lo[i]; r[i + 8] = hi[i]; }
    return r;
}
__device__ __forceinline__ v8f wmma_f16(v16h a, v16h b, v8f c) {
    return __builtin_amdgcn_wmma_f32_16x16x32_f16(false, a, false, b, (short)0, c,
                                                  false, false);
}

// ---------------------------------------------------------------------------
// Kernel 4: GEMM1  z = relu(p[4096x2048] @ EW[2048x256] + b1), z stored f16.
// 32x32 tile per wave: 4 accumulators, A/B fragments reused 2x.
// ---------------------------------------------------------------------------
__global__ __launch_bounds__(256) void k_gemm1(const _Float16* __restrict__ pbuf,
                                               const _Float16* __restrict__ ewt,
                                               const float* __restrict__ bias,
                                               _Float16* __restrict__ z) {
    const int wid  = blockIdx.x * 8 + (threadIdx.x >> 5);
    const int lane = threadIdx.x & 31;
    const int kh   = lane >> 4;
    const int idx  = lane & 15;
    const int mt = wid >> 3;          // 128 M-tiles
    const int nt = wid & 7;           // 8  N-tiles
    const int m0 = mt * 32, n0 = nt * 32;

    const _Float16* a0p = pbuf + (size_t)(m0 + idx) * KP;
    const _Float16* a1p = pbuf + (size_t)(m0 + 16 + idx) * KP;
    const _Float16* b0p = ewt  + (size_t)(n0 + idx) * KP;
    const _Float16* b1p = ewt  + (size_t)(n0 + 16 + idx) * KP;

    v8f acc00 = {}, acc01 = {}, acc10 = {}, acc11 = {};
#pragma unroll 2
    for (int k0 = 0; k0 < KP; k0 += 32) {
        v16h a0 = ld_frag(a0p, k0, kh);
        v16h a1 = ld_frag(a1p, k0, kh);
        v16h b0 = ld_frag(b0p, k0, kh);
        v16h b1 = ld_frag(b1p, k0, kh);
        acc00 = wmma_f16(a0, b0, acc00);
        acc01 = wmma_f16(a0, b1, acc01);
        acc10 = wmma_f16(a1, b0, acc10);
        acc11 = wmma_f16(a1, b1, acc11);
    }
    const float bias0 = bias[n0 + idx];
    const float bias1 = bias[n0 + 16 + idx];
#pragma unroll
    for (int r = 0; r < 8; ++r) {
        const int mA = m0 + kh * 8 + r;
        const int mB = mA + 16;
        z[(size_t)mA * NZ + n0 + idx]      = (_Float16)fmaxf(acc00[r] + bias0, 0.f);
        z[(size_t)mA * NZ + n0 + 16 + idx] = (_Float16)fmaxf(acc01[r] + bias1, 0.f);
        z[(size_t)mB * NZ + n0 + idx]      = (_Float16)fmaxf(acc10[r] + bias0, 0.f);
        z[(size_t)mB * NZ + n0 + 16 + idx] = (_Float16)fmaxf(acc11[r] + bias1, 0.f);
    }
}

// ---------------------------------------------------------------------------
// Kernel 5: GEMM2  out = z[4096x256] @ W2[256x4096] + b2 (f32 output).
// ---------------------------------------------------------------------------
__global__ __launch_bounds__(256) void k_gemm2(const _Float16* __restrict__ z,
                                               const _Float16* __restrict__ w2t,
                                               const float* __restrict__ bias,
                                               float* __restrict__ out) {
    const int wid  = blockIdx.x * 8 + (threadIdx.x >> 5);
    const int lane = threadIdx.x & 31;
    const int kh   = lane >> 4;
    const int idx  = lane & 15;
    const int mt = wid >> 7;          // 128 M-tiles
    const int nt = wid & 127;         // 128 N-tiles
    const int m0 = mt * 32, n0 = nt * 32;

    const _Float16* a0p = z   + (size_t)(m0 + idx) * NZ;
    const _Float16* a1p = z   + (size_t)(m0 + 16 + idx) * NZ;
    const _Float16* b0p = w2t + (size_t)(n0 + idx) * NZ;
    const _Float16* b1p = w2t + (size_t)(n0 + 16 + idx) * NZ;

    v8f acc00 = {}, acc01 = {}, acc10 = {}, acc11 = {};
#pragma unroll
    for (int k0 = 0; k0 < NZ; k0 += 32) {
        v16h a0 = ld_frag(a0p, k0, kh);
        v16h a1 = ld_frag(a1p, k0, kh);
        v16h b0 = ld_frag(b0p, k0, kh);
        v16h b1 = ld_frag(b1p, k0, kh);
        acc00 = wmma_f16(a0, b0, acc00);
        acc01 = wmma_f16(a0, b1, acc01);
        acc10 = wmma_f16(a1, b0, acc10);
        acc11 = wmma_f16(a1, b1, acc11);
    }
    const float bias0 = bias[n0 + idx];
    const float bias1 = bias[n0 + 16 + idx];
#pragma unroll
    for (int r = 0; r < 8; ++r) {
        const int mA = m0 + kh * 8 + r;
        const int mB = mA + 16;
        out[(size_t)mA * LLMW + n0 + idx]      = acc00[r] + bias0;
        out[(size_t)mA * LLMW + n0 + 16 + idx] = acc01[r] + bias1;
        out[(size_t)mB * LLMW + n0 + idx]      = acc10[r] + bias0;
        out[(size_t)mB * LLMW + n0 + 16 + idx] = acc11[r] + bias1;
    }
}

// ---------------------------------------------------------------------------
extern "C" void kernel_launch(void* const* d_in, const int* in_sizes, int n_in,
                              void* d_out, int out_size, void* d_ws, size_t ws_size,
                              hipStream_t stream) {
    const float* stat = (const float*)d_in[0];
    const float* l1w  = (const float*)d_in[1];
    const float* l1b  = (const float*)d_in[2];
    const float* l2w  = (const float*)d_in[3];
    const float* l2b  = (const float*)d_in[4];
    const float* tau  = (const float*)d_in[5];
    const float* emb  = (const float*)d_in[6];
    const float* pw1  = (const float*)d_in[7];
    const float* pb1  = (const float*)d_in[8];
    const float* pw2  = (const float*)d_in[9];
    const float* pb2  = (const float*)d_in[10];
    float* out = (float*)d_out;

    // workspace layout (f16 tensors, 256B aligned):
    //   pbuf : [4096][2048]  16 MiB
    //   ewt  : [256][2048]    1 MiB
    //   zbuf : [4096][256]    2 MiB
    //   w2t  : [4096][256]    2 MiB
    char* ws = (char*)d_ws;
    _Float16* pbuf = (_Float16*)(ws);
    _Float16* ewt  = (_Float16*)(ws + (size_t)16 * 1024 * 1024);
    _Float16* zbuf = (_Float16*)(ws + (size_t)17 * 1024 * 1024);
    _Float16* w2t  = (_Float16*)(ws + (size_t)19 * 1024 * 1024);

    k_ew_prep<<<dim3(F_FEAT), dim3(256), 0, stream>>>(emb, pw1, ewt);
    k_w2t    <<<dim3(LLMW / 32, NZ / 32), dim3(32, 8), 0, stream>>>(pw2, w2t);
    k_probs  <<<dim3(F_FEAT), dim3(256), 0, stream>>>(stat, l1w, l1b, l2w, l2b, tau, pbuf);
    k_gemm1  <<<dim3(128),  dim3(256), 0, stream>>>(pbuf, ewt, pb1, zbuf);
    k_gemm2  <<<dim3(2048), dim3(256), 0, stream>>>(zbuf, w2t, pb2, out);
}
